// MBTFEConv_38328288150258
// MI455X (gfx1250) — matrix operand order repaired and checked by reading, not compiled, not run on hardware
//
#include <hip/hip_runtime.h>
#include <math.h>

#define NN   50000
#define EE   800000
#define FDIM 128
#define NF   (NN * FDIM)        // 6,400,000 floats per feature buffer
#define KORD 8

typedef __attribute__((ext_vector_type(16))) __bf16        v16bf;
typedef __attribute__((ext_vector_type(8)))  float         v8f;
typedef __attribute__((ext_vector_type(4)))  unsigned int  v4u;

union BF16Frag { v16bf v; v4u u[2]; };

__device__ __forceinline__ unsigned short f2bf(float f) {
  unsigned int u = __float_as_uint(f);
  u += 0x7fffu + ((u >> 16) & 1u);          // round-to-nearest-even
  return (unsigned short)(u >> 16);
}

// Fragment builders matching CDNA5 16-bit WMMA VGPR layouts (two b128 each).
__device__ __forceinline__ void load_a_frag(BF16Frag& f,
                                            const unsigned short* row,
                                            int kt, int half) {
  int ka = kt * 32 + half * 8;              // A 16x32: K runs {0..7,16..23}/{8..15,24..31}
  f.u[0] = *(const v4u*)(row + ka);
  f.u[1] = *(const v4u*)(row + ka + 16);
}
__device__ __forceinline__ void load_b_frag(BF16Frag& f,
                                            const unsigned short* row,
                                            int kt, int half) {
  int kb = kt * 32 + half * 8;              // B 32x16: K runs {0..7,16..23}/{8..15,24..31}
  f.u[0] = *(const v4u*)(row + kb);
  f.u[1] = *(const v4u*)(row + kb + 16);
}

// ---------------------------------------------------------------------------
// bands[i] = C[i][0] * X   (k = 0 Chebyshev term; Psi_0 = X)
// ---------------------------------------------------------------------------
__global__ __launch_bounds__(256) void init_bands_kernel(
    const float* __restrict__ X, float* __restrict__ bands, float4 c) {
  int idx = blockIdx.x * blockDim.x + threadIdx.x;
  if (idx >= NF / 4) return;
  float4 x = reinterpret_cast<const float4*>(X)[idx];
  float cs[4] = {c.x, c.y, c.z, c.w};
#pragma unroll
  for (int i = 0; i < 4; ++i) {
    float4 v;
    v.x = cs[i] * x.x; v.y = cs[i] * x.y; v.z = cs[i] * x.z; v.w = cs[i] * x.w;
    reinterpret_cast<float4*>(bands + (size_t)i * NF)[idx] = v;
  }
}

// ---------------------------------------------------------------------------
// S[row] += w * src[col] : one wave per edge, lane owns 4 features (float4)
// ---------------------------------------------------------------------------
__global__ __launch_bounds__(256) void spmm_kernel(
    const int* __restrict__ row, const int* __restrict__ col,
    const float* __restrict__ w, const float* __restrict__ src,
    float* __restrict__ dst) {
  int gid  = blockIdx.x * blockDim.x + threadIdx.x;
  int e    = gid >> 5;
  int lane = gid & 31;
  if (e >= EE) return;
  int   r  = row[e];
  int   c  = col[e];
  float wt = w[e];
  float4 v = reinterpret_cast<const float4*>(src + (size_t)c * FDIM)[lane];
  float* d = dst + (size_t)r * FDIM + lane * 4;
  unsafeAtomicAdd(d + 0, wt * v.x);
  unsafeAtomicAdd(d + 1, wt * v.y);
  unsafeAtomicAdd(d + 2, wt * v.z);
  unsafeAtomicAdd(d + 3, wt * v.w);
}

// ---------------------------------------------------------------------------
// Pc = a*S + b*P2  (Chebyshev recurrence), then bands[i] += C[i][k] * Pc
// ---------------------------------------------------------------------------
__global__ __launch_bounds__(256) void cheb_combine_kernel(
    const float* __restrict__ S, const float* __restrict__ P2,
    float* __restrict__ Pc, float* __restrict__ bands,
    float a, float b, float4 c) {
  int idx = blockIdx.x * blockDim.x + threadIdx.x;
  if (idx >= NF / 4) return;
  float4 s = reinterpret_cast<const float4*>(S)[idx];
  float4 p = reinterpret_cast<const float4*>(P2)[idx];
  float4 pc;
  pc.x = a * s.x + b * p.x; pc.y = a * s.y + b * p.y;
  pc.z = a * s.z + b * p.z; pc.w = a * s.w + b * p.w;
  reinterpret_cast<float4*>(Pc)[idx] = pc;
  float cs[4] = {c.x, c.y, c.z, c.w};
#pragma unroll
  for (int i = 0; i < 4; ++i) {
    float4* bp = reinterpret_cast<float4*>(bands + (size_t)i * NF) + idx;
    float4 bv = *bp;
    bv.x += cs[i] * pc.x; bv.y += cs[i] * pc.y;
    bv.z += cs[i] * pc.z; bv.w += cs[i] * pc.w;
    *bp = bv;
  }
}

// ---------------------------------------------------------------------------
// Band GEMM: H_i = relu(band_i @ W_band[i] + b_band[i]) -> Hcat[:, i*128:+128]
// and Hcat[:, 512:640] = bf16(X).   32 nodes / block, 8 waves: (band, half).
// All 8 B-fragment ds_load_b128s issued before the 4-WMMA chain per col-tile.
// ---------------------------------------------------------------------------
__global__ __launch_bounds__(256) void band_gemm_kernel(
    const float* __restrict__ bands, const float* __restrict__ X,
    const float* __restrict__ Wb, const float* __restrict__ bb,
    unsigned short* __restrict__ Hcat) {
  extern __shared__ char smem[];
  unsigned short* ldsW  = (unsigned short*)smem;       // [4][128][128] (o-major)
  unsigned short* ldsA  = ldsW + 4 * 128 * 128;        // [4][32][128]
  float*          ldsBi = (float*)(ldsA + 4 * 32 * 128);  // [4][128]
  int tid   = threadIdx.x;
  int node0 = blockIdx.x * 32;
  bool full = (node0 + 32 <= NN);

  // stage W_band: Wb[i][f][o] -> ldsW[i][o][f], f32 -> bf16
  for (int e = tid; e < 4 * 128 * 128; e += 256) {
    int i = e >> 14, rem = e & 16383, o = rem >> 7, f = rem & 127;
    ldsW[e] = f2bf(Wb[i * 16384 + f * 128 + o]);
  }
  // stage band tiles (rows clamped for the tail block)
  for (int e = tid; e < 4 * 32 * 128; e += 256) {
    int i = e >> 12, rem = e & 4095, r = rem >> 7, f = rem & 127;
    int n = node0 + r; if (n > NN - 1) n = NN - 1;
    ldsA[e] = f2bf(bands[(size_t)i * NF + (size_t)n * FDIM + f]);
  }
  if (tid < 512 / 2) {   // 256 threads, 512 bias floats -> 2 each
    ldsBi[tid]       = bb[tid];
    ldsBi[tid + 256] = bb[tid + 256];
  }
  __syncthreads();

  int wave = tid >> 5, lane = tid & 31;
  int band = wave & 3, h = wave >> 2;       // h: which 16-node half
  int m = lane & 15, half = lane >> 4;
  const unsigned short* Arow = ldsA + band * 4096 + (h * 16 + m) * 128;
  const unsigned short* Wbas = ldsW + band * 16384;

  BF16Frag af[4];
#pragma unroll
  for (int kt = 0; kt < 4; ++kt) load_a_frag(af[kt], Arow, kt, half);

  for (int ct = 0; ct < 8; ++ct) {
    const unsigned short* Brow = Wbas + (ct * 16 + m) * 128;
    BF16Frag bfr[4];
#pragma unroll
    for (int kt = 0; kt < 4; ++kt) load_b_frag(bfr[kt], Brow, kt, half);
    v8f acc = {};
#pragma unroll
    for (int kt = 0; kt < 4; ++kt)
      acc = __builtin_amdgcn_wmma_f32_16x16x32_bf16(
          false, af[kt].v, false, bfr[kt].v, (short)0, acc, false, false);

    int o = ct * 16 + m;
    float bias = ldsBi[band * 128 + o];
    unsigned short* dstc = Hcat + (size_t)(node0 + h * 16) * 640 +
                           band * 128 + o;
    if (full) {
#pragma unroll
      for (int r = 0; r < 8; ++r) {             // C/D: M = r + 8*half, N = m
        float v = acc[r] + bias;
        v = v > 0.f ? v : 0.f;
        dstc[(size_t)(r + half * 8) * 640] = f2bf(v);
      }
    } else {
#pragma unroll
      for (int r = 0; r < 8; ++r) {
        int n = node0 + h * 16 + r + half * 8;
        float v = acc[r] + bias;
        v = v > 0.f ? v : 0.f;
        if (n < NN) Hcat[(size_t)n * 640 + band * 128 + o] = f2bf(v);
      }
    }
  }
  // Hcat[:, 512:640] = bf16(X)
  if (full) {
    for (int e = tid; e < 32 * 128; e += 256) {
      int r = e >> 7, f = e & 127, n = node0 + r;
      Hcat[(size_t)n * 640 + 512 + f] = f2bf(X[(size_t)n * FDIM + f]);
    }
  } else {
    for (int e = tid; e < 32 * 128; e += 256) {
      int r = e >> 7, f = e & 127, n = node0 + r;
      if (n < NN)
        Hcat[(size_t)n * 640 + 512 + f] = f2bf(X[(size_t)n * FDIM + f]);
    }
  }
}

// ---------------------------------------------------------------------------
// Fuse GEMM: out = Hcat[N,640] @ W_fuse[640,128] + b_fuse   (bf16 WMMA, f32 D)
// 32 nodes / block; double-buffered fragment pipeline over 20 k-steps.
// ---------------------------------------------------------------------------
__global__ __launch_bounds__(256) void fuse_gemm_kernel(
    const unsigned short* __restrict__ Hcat, const float* __restrict__ Wf,
    const float* __restrict__ bf, float* __restrict__ out) {
  extern __shared__ char smem[];
  unsigned short* ldsW  = (unsigned short*)smem;    // [128 out][640 in]
  unsigned short* ldsA  = ldsW + 128 * 640;         // [32][640]
  float*          ldsBi = (float*)(ldsA + 32 * 640);   // [128]
  int tid   = threadIdx.x;
  int node0 = blockIdx.x * 32;
  bool full = (node0 + 32 <= NN);

  for (int e = tid; e < 128 * 640; e += 256) {     // Wf[k][o] -> ldsW[o][k]
    int o = e / 640, k = e - o * 640;
    ldsW[e] = f2bf(Wf[k * 128 + o]);
  }
  for (int e = tid; e < 32 * 80; e += 256) {       // tile copy, 16B chunks
    int r = e / 80, cg = e - r * 80;
    int n = node0 + r; if (n > NN - 1) n = NN - 1;
    reinterpret_cast<v4u*>(ldsA)[r * 80 + cg] =
        reinterpret_cast<const v4u*>(Hcat + (size_t)n * 640)[cg];
  }
  if (tid < 128) ldsBi[tid] = bf[tid];
  __syncthreads();

  int wave = tid >> 5, lane = tid & 31;
  int m = lane & 15, half = lane >> 4;
  int h = wave >> 2;                                // node half
  const unsigned short* Arow = ldsA + (h * 16 + m) * 640;

  for (int cc = 0; cc < 2; ++cc) {
    int ct = (wave & 3) + cc * 4;
    const unsigned short* Brow = ldsW + (ct * 16 + m) * 640;
    BF16Frag a0, b0, a1, b1;
    load_a_frag(a0, Arow, 0, half);
    load_b_frag(b0, Brow, 0, half);
    v8f acc = {};
#pragma unroll
    for (int kt = 0; kt < 20; ++kt) {
      if (kt < 19) {                      // prefetch next fragments
        load_a_frag(a1, Arow, kt + 1, half);
        load_b_frag(b1, Brow, kt + 1, half);
      }
      acc = __builtin_amdgcn_wmma_f32_16x16x32_bf16(
          false, a0.v, false, b0.v, (short)0, acc, false, false);
      a0 = a1;
      b0 = b1;
    }

    int o = ct * 16 + m;
    float bias = ldsBi[o];
    float* dstc = out + (size_t)(node0 + h * 16) * FDIM + o;
    if (full) {
#pragma unroll
      for (int r = 0; r < 8; ++r)
        dstc[(size_t)(r + half * 8) * FDIM] = acc[r] + bias;
    } else {
#pragma unroll
      for (int r = 0; r < 8; ++r) {
        int n = node0 + h * 16 + r + half * 8;
        if (n < NN) out[(size_t)n * FDIM + o] = acc[r] + bias;
      }
    }
  }
}

// ---------------------------------------------------------------------------
// Host side
// ---------------------------------------------------------------------------
static double iv_host(int k, double tau) {  // modified Bessel I_k, power series
  double x = tau * 0.5;
  double t = 1.0;
  for (int i = 1; i <= k; ++i) t *= x / (double)i;   // x^k / k!
  double s = t;
  for (int mm = 0; mm < 60; ++mm) {
    t *= x * x / ((mm + 1.0) * (mm + 1.0 + k));
    s += t;
  }
  return s;
}

extern "C" void kernel_launch(void* const* d_in, const int* in_sizes, int n_in,
                              void* d_out, int out_size, void* d_ws, size_t ws_size,
                              hipStream_t stream) {
  const float* X   = (const float*)d_in[0];
  const int*   er  = (const int*)d_in[1];
  const int*   ec  = (const int*)d_in[2];
  const float* ew  = (const float*)d_in[3];
  const float* Wb  = (const float*)d_in[4];
  const float* bbn = (const float*)d_in[5];
  const float* Wf  = (const float*)d_in[6];
  const float* bfu = (const float*)d_in[7];

  // Static Chebyshev heat-kernel coefficients (taus fixed) -> band coeffs C.
  const double taus[4] = {0.0, 0.5, 1.5, 4.0};
  double A[4][KORD + 1];
  for (int i = 0; i < 4; ++i) {
    double tau = taus[i];
    if (tau == 0.0) {
      A[i][0] = 1.0;
      for (int k = 1; k <= KORD; ++k) A[i][k] = 0.0;
    } else {
      double e = exp(-tau);
      for (int k = 0; k <= KORD; ++k) {
        double Ik = iv_host(k, tau);
        A[i][k] = (k == 0) ? e * Ik : 2.0 * e * ((k & 1) ? -1.0 : 1.0) * Ik;
      }
    }
  }
  float C[4][KORD + 1];
  for (int k = 0; k <= KORD; ++k) {
    for (int i = 0; i < 3; ++i) C[i][k] = (float)(A[i][k] - A[i + 1][k]);
    C[3][k] = (float)A[3][k];
  }

  // Workspace layout (floats): bands[4*NF] | psiA | psiB | S ; Hcat (bf16,
  // N*640 = 64MB) overlaps the psiA..S region (76.8MB) once Chebyshev is done.
  float* ws    = (float*)d_ws;
  float* bands = ws;
  float* psiA  = ws + (size_t)4 * NF;
  float* psiB  = ws + (size_t)5 * NF;
  float* S     = ws + (size_t)6 * NF;
  unsigned short* Hcat = (unsigned short*)(ws + (size_t)4 * NF);

  const int combBlocks = (NF / 4) / 256;   // 6250
  const int spmmBlocks = EE / 8;           // 100000 (8 edge-waves per block)
  const int gemmBlocks = (NN + 31) / 32;   // 1563

  size_t bandShmem =
      (size_t)(4 * 128 * 128 + 4 * 32 * 128) * 2 + 512 * sizeof(float);
  size_t fuseShmem =
      (size_t)(128 * 640 + 32 * 640) * 2 + 128 * sizeof(float);
  hipFuncSetAttribute((const void*)band_gemm_kernel,
                      hipFuncAttributeMaxDynamicSharedMemorySize, (int)bandShmem);
  hipFuncSetAttribute((const void*)fuse_gemm_kernel,
                      hipFuncAttributeMaxDynamicSharedMemorySize, (int)fuseShmem);

  // k = 0 term
  float4 c0 = make_float4(C[0][0], C[1][0], C[2][0], C[3][0]);
  init_bands_kernel<<<combBlocks, 256, 0, stream>>>(X, bands, c0);

  // k = 1: Psi_1 = L X  (into psiA)
  hipMemsetAsync(psiA, 0, sizeof(float) * (size_t)NF, stream);
  spmm_kernel<<<spmmBlocks, 256, 0, stream>>>(er, ec, ew, X, psiA);
  float4 c1 = make_float4(C[0][1], C[1][1], C[2][1], C[3][1]);
  cheb_combine_kernel<<<combBlocks, 256, 0, stream>>>(psiA, X, psiA, bands,
                                                      1.f, 0.f, c1);

  // k = 2..8: Psi_k = 2 L Psi_{k-1} - Psi_{k-2}  (ping-pong psiA/psiB)
  for (int k = 2; k <= KORD; ++k) {
    const float* src = (k & 1) ? psiB : psiA;                   // Psi_{k-1}
    const float* p2  = (k == 2) ? X : ((k & 1) ? psiA : psiB);  // Psi_{k-2}
    float*       pc  = (k & 1) ? psiA : psiB;                   // Psi_k
    hipMemsetAsync(S, 0, sizeof(float) * (size_t)NF, stream);
    spmm_kernel<<<spmmBlocks, 256, 0, stream>>>(er, ec, ew, src, S);
    float4 ck = make_float4(C[0][k], C[1][k], C[2][k], C[3][k]);
    cheb_combine_kernel<<<combBlocks, 256, 0, stream>>>(S, p2, pc, bands,
                                                        2.f, -1.f, ck);
  }

  // Dense phase: WMMA bf16 GEMMs
  band_gemm_kernel<<<gemmBlocks, 256, bandShmem, stream>>>(bands, X, Wb, bbn,
                                                           Hcat);
  fuse_gemm_kernel<<<gemmBlocks, 256, fuseShmem, stream>>>(Hcat, Wf, bfu,
                                                           (float*)d_out);
}
